// SparseAutoencoder_91096256348512
// MI455X (gfx1250) — matrix operand order repaired and checked by reading, not compiled, not run on hardware
//
#include <hip/hip_runtime.h>

#define BATCH   262144
#define IN_DIM  64
#define LAT     512
#define TOPK    32
#define TM      64          // rows per block
#define THREADS 256         // 8 waves (wave32)
#define ZPAD    516         // padded z row stride (floats) -> bank-conflict free, 16B-aligned rows

#define AS1 __attribute__((address_space(1)))
#define AS3 __attribute__((address_space(3)))

#if __has_builtin(__builtin_amdgcn_global_load_async_to_lds_b128) && \
    __has_builtin(__builtin_amdgcn_global_store_async_from_lds_b128) && \
    __has_builtin(__builtin_amdgcn_s_wait_asynccnt)
#define HAS_ASYNC_LDS 1
#else
#define HAS_ASYNC_LDS 0
#endif

typedef int v4i __attribute__((vector_size(16)));               // matches builtin param type
#define GPTR128(p) ((AS1 v4i*)(unsigned long long)(const void*)(p))
#define LPTR128(p) ((AS3 v4i*)(p))

typedef __attribute__((ext_vector_type(16))) _Float16 v16h;
typedef __attribute__((ext_vector_type(8)))  float    v8f;

__device__ __forceinline__ int imax(int a, int b) { return a > b ? a : b; }

__global__ __launch_bounds__(THREADS)
void sae_fused_kernel(const float* __restrict__ x,
                      const float* __restrict__ encW,   // [512,64]
                      const float* __restrict__ encB,   // [512]
                      const float* __restrict__ decW,   // [64,512]
                      const float* __restrict__ decB,   // [64]
                      float* __restrict__ out)          // [B*64] x_hat ++ [B*512] z_sparse
{
    __shared__ float xF[TM * IN_DIM];   // 16 KB  : x tile (f32)
    __shared__ float zs[TM * ZPAD];     // 129 KB : z tile (f32, padded rows)

    const int tid  = threadIdx.x;
    const int lane = tid & 31;
    const int wave = tid >> 5;
    const int rowBase = blockIdx.x * TM;

    // Warm L2 with decoder weights (used in phase 3; 128KB stays L2-hot across blocks)
    __builtin_prefetch(&decW[tid * 128], 0, 0);

    // ---- phase 0: stage x tile to LDS ----
#if HAS_ASYNC_LDS
    for (int i = tid * 4; i < TM * IN_DIM; i += THREADS * 4)
        __builtin_amdgcn_global_load_async_to_lds_b128(
            GPTR128(x + (size_t)rowBase * IN_DIM + i), LPTR128(xF + i), 0, 0);
    __builtin_amdgcn_s_wait_asynccnt(0);
#else
    for (int i = tid; i < TM * IN_DIM; i += THREADS)
        xF[i] = x[(size_t)rowBase * IN_DIM + i];
#endif
    __syncthreads();

    // lane-derived WMMA fragment coordinates (16-bit A 16x32 / B 32x16; contiguous k = 16*hi + j in A and B, not the ISA table order)
    const int l15   = lane & 15;
    const int hi    = lane >> 4;       // 0: lanes 0-15, 1: lanes 16-31
    const int kbase = hi ? 16 : 0;     // A-frag K sub-base
    const int klo   = hi ? 16 : 0;     // B-frag K sub-base

    // ---- phase 1: encoder GEMM  z = relu(x @ encW^T + encB) ----
    {
        const int mb   = wave & 3;            // 4 M-blocks of 16
        const int nb0  = (wave >> 2) * 16;    // 2 halves of 32 N-blocks
        const int mrow = mb * 16 + l15;
        v16h a0, a1;
        #pragma unroll
        for (int j = 0; j < 16; ++j) {
            const int k = kbase + j;
            a0[j] = (_Float16)xF[mrow * IN_DIM + k];
            a1[j] = (_Float16)xF[mrow * IN_DIM + 32 + k];
        }
        for (int nb = nb0; nb < nb0 + 16; ++nb) {
            const int n = nb * 16 + l15;
            v16h b0, b1;
            #pragma unroll
            for (int j = 0; j < 16; ++j) {            // per-lane contiguous encW row reads
                b0[j] = (_Float16)encW[n * IN_DIM + klo + j];
                b1[j] = (_Float16)encW[n * IN_DIM + 32 + klo + j];
            }
            const float bias = encB[n];               // C cols == latent index
            v8f c;
            #pragma unroll
            for (int r = 0; r < 8; ++r) c[r] = bias;
            c = __builtin_amdgcn_wmma_f32_16x16x32_f16(false, a0, false, b0, (short)0, c, false, false);
            c = __builtin_amdgcn_wmma_f32_16x16x32_f16(false, a1, false, b1, (short)0, c, false, false);
            const int mtop = mb * 16 + hi * 8;
            #pragma unroll
            for (int r = 0; r < 8; ++r)
                zs[(mtop + r) * ZPAD + n] = fmaxf(c[r], 0.0f);   // ReLU
        }
    }
    __syncthreads();

    // ---- phase 2: per-row top-K; sparsify zs in place ----
    {
        for (int rr = 0; rr < TM / 8; ++rr) {
            const int row = wave * (TM / 8) + rr;
            float v[16];
            int   key[16];
            #pragma unroll
            for (int i = 0; i < 16; ++i) {            // lane owns latents {lane, lane+32, ...}
                v[i] = zs[row * ZPAD + i * 32 + lane];
                // ReLU => v>=0 => float bits are order-preserving; pack 9-bit latent id into mantissa LSBs
                key[i] = (int)((__float_as_uint(v[i]) & 0xFFFFFE00u) | (unsigned)(i * 32 + lane));
            }
            unsigned used = 0;
            int lbest = 0;
            #pragma unroll
            for (int i = 0; i < 16; ++i) lbest = imax(lbest, key[i]);
            for (int t = 0; t < TOPK; ++t) {
                int k = lbest;
                #pragma unroll
                for (int off = 16; off >= 1; off >>= 1)
                    k = imax(k, __shfl_xor(k, off, 32));   // wave max over packed keys
                const int latent = k & 511;
                if ((latent & 31) == lane) {               // unique winner lane pops + rescans
                    used |= 1u << (latent >> 5);
                    lbest = 0;
                    #pragma unroll
                    for (int i = 0; i < 16; ++i)
                        if (!(used & (1u << i))) lbest = imax(lbest, key[i]);
                }
            }
            #pragma unroll
            for (int i = 0; i < 16; ++i) {
                const float zv = (used & (1u << i)) ? v[i] : 0.0f;
                zs[row * ZPAD + i * 32 + lane] = zv;   // feed decoder GEMM + async stream-out
#if !HAS_ASYNC_LDS
                __builtin_nontemporal_store(zv,
                    out + (size_t)BATCH * IN_DIM + (size_t)(rowBase + row) * LAT + i * 32 + lane);
#endif
            }
        }
    }
    __syncthreads();

#if HAS_ASYNC_LDS
    // ---- stream z_sparse straight out of LDS (ASYNCcnt); overlaps decoder WMMA below ----
    {
        float* zOutBase = out + (size_t)BATCH * IN_DIM + (size_t)rowBase * LAT;
        for (int c = tid; c < TM * (LAT / 4); c += THREADS) {
            const int row = c >> 7;            // 128 x b128 chunks per row
            const int col = (c & 127) * 4;
            __builtin_amdgcn_global_store_async_from_lds_b128(
                GPTR128(zOutBase + (size_t)row * LAT + col),
                LPTR128(zs + row * ZPAD + col), 0, 0);
        }
    }
#endif

    // ---- phase 3: decoder GEMM  x_hat = z_sparse @ decW^T + decB ----
    {
        for (int t = wave * 2; t < wave * 2 + 2; ++t) {   // 16 tiles (4x4), 2 per wave
            const int mb = t >> 2, nb = t & 3;
            const int mrow = mb * 16 + l15;
            const int ncol = nb * 16 + l15;
            const float bias = decB[ncol];
            v8f c;
            #pragma unroll
            for (int r = 0; r < 8; ++r) c[r] = bias;
            for (int ko = 0; ko < LAT; ko += 32) {
                v16h a, b;
                #pragma unroll
                for (int j = 0; j < 16; ++j) {
                    const int k = kbase + j;
                    a[j] = (_Float16)zs[mrow * ZPAD + ko + k];
                    b[j] = (_Float16)decW[ncol * LAT + ko + klo + j];  // K-contiguous
                }
                c = __builtin_amdgcn_wmma_f32_16x16x32_f16(false, a, false, b, (short)0, c, false, false);
            }
            const int mtop = mb * 16 + hi * 8;
            #pragma unroll
            for (int r = 0; r < 8; ++r)
                __builtin_nontemporal_store(c[r],
                    out + (size_t)(rowBase + mtop + r) * IN_DIM + ncol);
        }
    }

#if HAS_ASYNC_LDS
    __builtin_amdgcn_s_wait_asynccnt(0);   // drain z_sparse stream before wave retire
#endif
}

extern "C" void kernel_launch(void* const* d_in, const int* in_sizes, int n_in,
                              void* d_out, int out_size, void* d_ws, size_t ws_size,
                              hipStream_t stream) {
    (void)in_sizes; (void)n_in; (void)out_size; (void)d_ws; (void)ws_size;
    const float* x    = (const float*)d_in[0];
    const float* encW = (const float*)d_in[1];
    const float* encB = (const float*)d_in[2];
    const float* decW = (const float*)d_in[3];
    const float* decB = (const float*)d_in[4];
    float* out = (float*)d_out;

    dim3 grid(BATCH / TM), block(THREADS);
    sae_fused_kernel<<<grid, block, 0, stream>>>(x, encW, encB, decW, decB, out);
}
